// AirQualityRecovery_46359876993743
// MI455X (gfx1250) — matrix operand order repaired and checked by reading, not compiled, not run on hardware
//
#include <hip/hip_runtime.h>
#include <hip/hip_bf16.h>
#include <math.h>

// ---------------------------------------------------------------------------
// Problem constants (match reference)
#define BATCH   128
#define TSTEPS  1024
#define FDIM    512
#define HDIM    512
#define G4      2048        // 4*H
#define KCAT    1024        // F + H (fused [x_t ; h_{t-1}] GEMM)
#define OUTC    24
#define MROWS   (BATCH * TSTEPS)   // 131072

// Persistent LSTM geometry: block = one h-tile (32 blocks), its 8 waves = 8 m-tiles
#define LSTM_BLOCKS 32
#define LSTM_THREADS 256
#define LSTM_LDS_BYTES (4 * 16 * KCAT * 2)   // 128 KiB weight slice per block

typedef __attribute__((ext_vector_type(16))) __bf16 v16bf;
typedef __attribute__((ext_vector_type(8)))  float  v8f;

union Frag16 { v16bf v; uint4 q[2]; };

__device__ __forceinline__ unsigned short f32_to_bf16(float f) {
    unsigned u = __float_as_uint(f);
    unsigned r = (u + 0x7FFFu + ((u >> 16) & 1u)) >> 16;
    return (unsigned short)r;
}

#define LOG2E 1.4426950408889634f

// Branch-free activations on native v_exp_f32 / v_rcp_f32 (no EXEC divergence)
__device__ __forceinline__ float fast_sigmoid(float x) {
    return __builtin_amdgcn_rcpf(1.0f + __builtin_amdgcn_exp2f(x * (-LOG2E)));
}
__device__ __forceinline__ float fast_tanh(float x) {
    // tanh(x) = 2*sigmoid(2x) - 1
    return 2.0f * __builtin_amdgcn_rcpf(1.0f + __builtin_amdgcn_exp2f(x * (-2.0f * LOG2E))) - 1.0f;
}

// ---------------------------------------------------------------------------
// Conversion / packing kernels
__global__ void k_convert_bf16(const float* __restrict__ in,
                               unsigned short* __restrict__ out, size_t n) {
    size_t i = (size_t)blockIdx.x * blockDim.x + threadIdx.x;
    size_t stride = (size_t)gridDim.x * blockDim.x;
    for (; i < n; i += stride) out[i] = f32_to_bf16(in[i]);
}

// Build WcatT[n][k] (n-major, n<2048, k<1024) = bf16 of [W;U] transposed.
__global__ void k_build_wcat(const float* __restrict__ W,   // [512,2048]
                             const float* __restrict__ U,   // [512,2048]
                             unsigned short* __restrict__ wT) {
    int i = blockIdx.x * blockDim.x + threadIdx.x;
    if (i >= G4 * KCAT) return;
    int n = i >> 10;          // 0..2047
    int k = i & 1023;         // 0..1023
    float v = (k < FDIM) ? W[(size_t)k * G4 + n] : U[(size_t)(k - FDIM) * G4 + n];
    wT[i] = f32_to_bf16(v);
}

// Weff = [Wh1|Wh2|Wh3] (512x24) @ Wf (24x24); stored transposed n-major,
// n padded to 32. beff = [bh1|bh2|bh3] @ Wf + bf.
__global__ void k_build_weff(const float* __restrict__ Wh1, const float* __restrict__ Wh2,
                             const float* __restrict__ Wh3, const float* __restrict__ bh1,
                             const float* __restrict__ bh2, const float* __restrict__ bh3,
                             const float* __restrict__ Wf,  const float* __restrict__ bfv,
                             unsigned short* __restrict__ weffT, float* __restrict__ beff) {
    int i = blockIdx.x * blockDim.x + threadIdx.x;   // 512*32 threads
    if (i >= HDIM * 32) return;
    int o = i & 31;           // padded out col
    int k = i >> 5;           // 0..511
    float acc = 0.0f;
    if (o < OUTC) {
        for (int j = 0; j < OUTC; ++j) {
            float w = (j < 8)  ? Wh1[k * 8 + j]
                    : (j < 16) ? Wh2[k * 8 + (j - 8)]
                               : Wh3[k * 8 + (j - 16)];
            acc += w * Wf[j * OUTC + o];
        }
    }
    weffT[(size_t)o * HDIM + k] = f32_to_bf16(acc);
    if (k == 0 && o < OUTC) {
        float b = bfv[o];
        for (int j = 0; j < OUTC; ++j) {
            float bj = (j < 8) ? bh1[j] : (j < 16) ? bh2[j - 8] : bh3[j - 16];
            b += bj * Wf[j * OUTC + o];
        }
        beff[o] = b;
    }
}

// ---------------------------------------------------------------------------
// Persistent fused LSTM layer.
//   xin : [B, T, 512] bf16 (layer input sequence)
//   hout: [B, T, 512] bf16 (layer output sequence; also recurrent input)
//   wT  : [2048][1024] bf16, n-major transposed [W;U]
//   bias: [2048] f32 (gate order i,f,g,o)
// Block = one 16-wide h-tile; its weight slice (128 KiB) is staged in LDS once.
// Each wave owns (mtile=waveid, htile=blockid); cell state lives in VGPRs.
__global__ __launch_bounds__(LSTM_THREADS)
void k_lstm_layer(const unsigned short* __restrict__ xin,
                  unsigned short* __restrict__ hout,
                  const unsigned short* __restrict__ wT,
                  const float* __restrict__ bias,
                  unsigned int* __restrict__ ctr) {
    extern __shared__ unsigned short lds_w[];   // [4 gates][16 cols][1024 k] bf16

    const int lane  = threadIdx.x & 31;
    const int wv    = threadIdx.x >> 5;
    const int htile = blockIdx.x;                             // 0..31
    const int mtile = wv;                                     // 0..7
    const int hh    = lane >> 4;                              // k-half / m-half
    const int lc    = lane & 15;
    const int mrow  = mtile * 16 + lc;                        // A-row (batch idx)
    const int ncol  = htile * 16 + lc;                        // C/D column (h idx)

    // --- Stage this block's weight slice into LDS (64 rows x 1024 bf16) ----
    {
        uint4* ldsq = (uint4*)lds_w;
        for (int idx = threadIdx.x; idx < LSTM_LDS_BYTES / 16; idx += LSTM_THREADS) {
            int row = idx >> 7;              // 0..63  (gate*16 + col)
            int u   = idx & 127;             // uint4 index within 1024 bf16 row
            int g   = row >> 4;
            int cc  = row & 15;
            ldsq[idx] = *((const uint4*)(wT + ((size_t)(g * HDIM + htile * 16 + cc)) * KCAT) + u);
        }
    }
    __syncthreads();

    const float bi = bias[0 * HDIM + ncol];
    const float bf = bias[1 * HDIM + ncol];
    const float bg = bias[2 * HDIM + ncol];
    const float bo = bias[3 * HDIM + ncol];

    const unsigned int nb = gridDim.x;
    v8f c = {};                                               // cell state tile

    for (int t = 0; t < TSTEPS; ++t) {
        v8f zi = {}, zf = {}, zg = {}, zo = {};

        const unsigned short* xrow = xin + ((size_t)mrow * TSTEPS + t) * FDIM;
        if (t + 1 < TSTEPS)
            __builtin_prefetch(xrow + FDIM, 0, 3);            // next step's x row

        // ---- x_t contribution: k-chunks 0..15 (LDS k offset = kbase) ------
        #pragma unroll 1
        for (int kc = 0; kc < FDIM / 32; ++kc) {
            const int kbase = kc * 32;
            Frag16 a;
            const unsigned short* src = xrow + kbase + hh * 16;
            a.q[0] = *(const uint4*)(src);
            a.q[1] = *(const uint4*)(src + 8);
            Frag16 b0, b1, b2, b3;
            const int kofs = kbase + hh * 16;
            const unsigned short* w0 = lds_w + (0 * 16 + lc) * KCAT + kofs;
            const unsigned short* w1 = lds_w + (1 * 16 + lc) * KCAT + kofs;
            const unsigned short* w2 = lds_w + (2 * 16 + lc) * KCAT + kofs;
            const unsigned short* w3 = lds_w + (3 * 16 + lc) * KCAT + kofs;
            b0.q[0] = *(const uint4*)(w0); b0.q[1] = *(const uint4*)(w0 + 8);
            b1.q[0] = *(const uint4*)(w1); b1.q[1] = *(const uint4*)(w1 + 8);
            b2.q[0] = *(const uint4*)(w2); b2.q[1] = *(const uint4*)(w2 + 8);
            b3.q[0] = *(const uint4*)(w3); b3.q[1] = *(const uint4*)(w3 + 8);
            zi = __builtin_amdgcn_wmma_f32_16x16x32_bf16(false, a.v, false, b0.v, (short)0, zi, false, false);
            zf = __builtin_amdgcn_wmma_f32_16x16x32_bf16(false, a.v, false, b1.v, (short)0, zf, false, false);
            zg = __builtin_amdgcn_wmma_f32_16x16x32_bf16(false, a.v, false, b2.v, (short)0, zg, false, false);
            zo = __builtin_amdgcn_wmma_f32_16x16x32_bf16(false, a.v, false, b3.v, (short)0, zo, false, false);
        }

        // ---- h_{t-1} contribution: k-chunks 16..31 (skip at t==0: h==0) ---
        if (t > 0) {
            const unsigned short* hrow = hout + ((size_t)mrow * TSTEPS + (t - 1)) * HDIM;
            #pragma unroll 1
            for (int kc = 0; kc < HDIM / 32; ++kc) {
                const int kbase = kc * 32;
                Frag16 a;
                const unsigned short* src = hrow + kbase + hh * 16;
                a.q[0] = *(const uint4*)(src);
                a.q[1] = *(const uint4*)(src + 8);
                Frag16 b0, b1, b2, b3;
                const int kofs = FDIM + kbase + hh * 16;      // h-part of K
                const unsigned short* w0 = lds_w + (0 * 16 + lc) * KCAT + kofs;
                const unsigned short* w1 = lds_w + (1 * 16 + lc) * KCAT + kofs;
                const unsigned short* w2 = lds_w + (2 * 16 + lc) * KCAT + kofs;
                const unsigned short* w3 = lds_w + (3 * 16 + lc) * KCAT + kofs;
                b0.q[0] = *(const uint4*)(w0); b0.q[1] = *(const uint4*)(w0 + 8);
                b1.q[0] = *(const uint4*)(w1); b1.q[1] = *(const uint4*)(w1 + 8);
                b2.q[0] = *(const uint4*)(w2); b2.q[1] = *(const uint4*)(w2 + 8);
                b3.q[0] = *(const uint4*)(w3); b3.q[1] = *(const uint4*)(w3 + 8);
                zi = __builtin_amdgcn_wmma_f32_16x16x32_bf16(false, a.v, false, b0.v, (short)0, zi, false, false);
                zf = __builtin_amdgcn_wmma_f32_16x16x32_bf16(false, a.v, false, b1.v, (short)0, zf, false, false);
                zg = __builtin_amdgcn_wmma_f32_16x16x32_bf16(false, a.v, false, b2.v, (short)0, zg, false, false);
                zo = __builtin_amdgcn_wmma_f32_16x16x32_bf16(false, a.v, false, b3.v, (short)0, zo, false, false);
            }
        }

        // gate nonlinearities + state update; C/D layout: row = r + 8*hh, col = ncol
        #pragma unroll
        for (int r = 0; r < 8; ++r) {
            float iv = fast_sigmoid(zi[r] + bi);
            float fv = fast_sigmoid(zf[r] + bf);
            float gv = fast_tanh(zg[r] + bg);
            float ov = fast_sigmoid(zo[r] + bo);
            float cv = fv * c[r] + iv * gv;
            c[r] = cv;
            float hv = ov * fast_tanh(cv);
            int brow = mtile * 16 + hh * 8 + r;
            hout[((size_t)brow * TSTEPS + t) * HDIM + ncol] = f32_to_bf16(hv);
        }

        // device-wide step barrier (monotonic counter, all blocks resident)
        __threadfence();
        __syncthreads();
        if (threadIdx.x == 0) {
            __hip_atomic_fetch_add(ctr, 1u, __ATOMIC_ACQ_REL, __HIP_MEMORY_SCOPE_AGENT);
            const unsigned int target = (unsigned int)(t + 1) * nb;
            while (__hip_atomic_load(ctr, __ATOMIC_ACQUIRE, __HIP_MEMORY_SCOPE_AGENT) < target)
                __builtin_amdgcn_s_sleep(8);
        }
        __syncthreads();
        __threadfence();
    }
}

// ---------------------------------------------------------------------------
// Fused heads + final linear: out[M,24] = h2[M,512] @ Weff + beff  (N padded 32)
__global__ __launch_bounds__(256)
void k_head_gemm(const unsigned short* __restrict__ h2,     // [M,512] bf16
                 const unsigned short* __restrict__ weffT,  // [32][512] bf16
                 const float* __restrict__ beff,            // [24]
                 float* __restrict__ out) {                 // [M,24]
    const int lane  = threadIdx.x & 31;
    const int wv    = threadIdx.x >> 5;
    const int mtile = blockIdx.x * 8 + wv;                  // 0..8191
    const int hh    = lane >> 4;
    const int lc    = lane & 15;
    const int mrow  = mtile * 16 + lc;

    v8f acc0 = {}, acc1 = {};
    #pragma unroll 1
    for (int kc = 0; kc < HDIM / 32; ++kc) {
        const int kbase = kc * 32;
        Frag16 a;
        const unsigned short* src = h2 + (size_t)mrow * HDIM + kbase + hh * 16;
        a.q[0] = *(const uint4*)(src);
        a.q[1] = *(const uint4*)(src + 8);
        Frag16 b0, b1;
        const unsigned short* w0 = weffT + (size_t)lc * HDIM + kbase + hh * 16;
        const unsigned short* w1 = weffT + (size_t)(16 + lc) * HDIM + kbase + hh * 16;
        b0.q[0] = *(const uint4*)(w0); b0.q[1] = *(const uint4*)(w0 + 8);
        b1.q[0] = *(const uint4*)(w1); b1.q[1] = *(const uint4*)(w1 + 8);
        acc0 = __builtin_amdgcn_wmma_f32_16x16x32_bf16(false, a.v, false, b0.v, (short)0, acc0, false, false);
        acc1 = __builtin_amdgcn_wmma_f32_16x16x32_bf16(false, a.v, false, b1.v, (short)0, acc1, false, false);
    }
    const float be0 = beff[lc];
    const float be1 = (lc < 8) ? beff[16 + lc] : 0.0f;
    #pragma unroll
    for (int r = 0; r < 8; ++r) {
        int row = mtile * 16 + hh * 8 + r;
        out[(size_t)row * OUTC + lc] = acc0[r] + be0;
        if (lc < 8) out[(size_t)row * OUTC + 16 + lc] = acc1[r] + be1;
    }
}

// ---------------------------------------------------------------------------
extern "C" void kernel_launch(void* const* d_in, const int* in_sizes, int n_in,
                              void* d_out, int out_size, void* d_ws, size_t ws_size,
                              hipStream_t stream) {
    (void)in_sizes; (void)n_in; (void)out_size; (void)ws_size;
    const float* x   = (const float*)d_in[0];
    const float* W1  = (const float*)d_in[1];
    const float* U1  = (const float*)d_in[2];
    const float* b1  = (const float*)d_in[3];
    const float* W2  = (const float*)d_in[4];
    const float* U2  = (const float*)d_in[5];
    const float* b2  = (const float*)d_in[6];
    const float* Wh1 = (const float*)d_in[7];
    const float* bh1 = (const float*)d_in[8];
    const float* Wh2 = (const float*)d_in[9];
    const float* bh2 = (const float*)d_in[10];
    const float* Wh3 = (const float*)d_in[11];
    const float* bh3 = (const float*)d_in[12];
    const float* Wf  = (const float*)d_in[13];
    const float* bfv = (const float*)d_in[14];
    float* out = (float*)d_out;

    // Workspace layout (all offsets 256B aligned)
    char* ws = (char*)d_ws;
    const size_t SZ_SEQ  = (size_t)BATCH * TSTEPS * HDIM * sizeof(unsigned short); // 128 MiB
    const size_t SZ_WCAT = (size_t)G4 * KCAT * sizeof(unsigned short);             // 4 MiB
    size_t off = 0;
    unsigned int*   ctrs  = (unsigned int*)(ws + off);  off += 256;
    unsigned short* xbf   = (unsigned short*)(ws + off); off += SZ_SEQ;
    unsigned short* h1bf  = (unsigned short*)(ws + off); off += SZ_SEQ;
    unsigned short* h2bf  = (unsigned short*)(ws + off); off += SZ_SEQ;
    unsigned short* wT1   = (unsigned short*)(ws + off); off += SZ_WCAT;
    unsigned short* wT2   = (unsigned short*)(ws + off); off += SZ_WCAT;
    unsigned short* weffT = (unsigned short*)(ws + off); off += (size_t)32 * HDIM * sizeof(unsigned short);
    float*          beff  = (float*)(ws + off);          off += 256;

    // Reset grid-barrier counters (capture-safe)
    hipMemsetAsync(ctrs, 0, 256, stream);

    // Pack inputs/weights to bf16
    const size_t nx = (size_t)BATCH * TSTEPS * FDIM;
    k_convert_bf16<<<8192, 256, 0, stream>>>(x, xbf, nx);
    k_build_wcat<<<(G4 * KCAT + 255) / 256, 256, 0, stream>>>(W1, U1, wT1);
    k_build_wcat<<<(G4 * KCAT + 255) / 256, 256, 0, stream>>>(W2, U2, wT2);
    k_build_weff<<<(HDIM * 32 + 255) / 256, 256, 0, stream>>>(
        Wh1, Wh2, Wh3, bh1, bh2, bh3, Wf, bfv, weffT, beff);

    // Two persistent LSTM layers (stream order serializes them)
    k_lstm_layer<<<LSTM_BLOCKS, LSTM_THREADS, LSTM_LDS_BYTES, stream>>>(xbf,  h1bf, wT1, b1, ctrs + 0);
    k_lstm_layer<<<LSTM_BLOCKS, LSTM_THREADS, LSTM_LDS_BYTES, stream>>>(h1bf, h2bf, wT2, b2, ctrs + 1);

    // Fused heads + final linear
    k_head_gemm<<<MROWS / 16 / 8, 256, 0, stream>>>(h2bf, weffT, beff, out);
}
